// FmriEncoder_EncMoE_77360950935860
// MI455X (gfx1250) — matrix operand order repaired and checked by reading, not compiled, not run on hardware
//
#include <hip/hip_runtime.h>

// ---------------- problem constants (from reference) ----------------
constexpr int HDIM    = 3072;   // hidden
constexpr int EXPERTS = 4;
constexpr int NTOK    = 8192;   // B*T = 4*2048
constexpr int LAYERS  = 2;

// ---------------- GEMM tiling ----------------
constexpr int TM = 128;         // tokens per block tile
constexpr int TN = 128;         // output cols per block tile
constexpr int TK = 32;          // k per wmma step (bf16 wmma K)
constexpr int LDS_STRIDE = 40;  // halfs per LDS row (32 + 8 pad -> conflict-free b128 frag reads)

typedef __bf16 v16bf __attribute__((ext_vector_type(16)));
typedef __bf16 v8bf  __attribute__((ext_vector_type(8)));
typedef __bf16 v4bf  __attribute__((ext_vector_type(4)));
typedef float  v8f   __attribute__((ext_vector_type(8)));

// gfx1250 async global->LDS copy path (guarded: falls back to ld+ds_store)
#if defined(__gfx1250__) &&                                            \
    __has_builtin(__builtin_amdgcn_global_load_async_to_lds_b128) &&   \
    __has_builtin(__builtin_amdgcn_s_wait_asynccnt)
#define USE_ASYNC_LDS 1
typedef int v4i __attribute__((ext_vector_type(4)));
typedef __attribute__((address_space(1))) v4i as1_v4i;   // global int4*
typedef __attribute__((address_space(3))) v4i as3_v4i;   // LDS int4*
#else
#define USE_ASYNC_LDS 0
#endif

// =====================================================================
// Kernel 1: We[e][h][d] fp32  ->  Wt[e][d][h] bf16 (transposed)
// So that a WMMA B-fragment column (contiguous K) is a contiguous row.
// =====================================================================
__global__ void __launch_bounds__(256)
cvt_transpose_bf16(const float* __restrict__ We, unsigned short* __restrict__ Wt_)
{
    __bf16* Wt = reinterpret_cast<__bf16*>(Wt_);
    __shared__ float s[32][33];
    const int e  = blockIdx.z;
    const int h0 = blockIdx.y * 32;
    const int d0 = blockIdx.x * 32;
    const int tx = threadIdx.x & 31;
    const int ty = threadIdx.x >> 5;   // 8 rows per pass, 4 passes

    const float* src = We + ((size_t)e * HDIM + h0) * HDIM + d0;
#pragma unroll
    for (int p = 0; p < 4; ++p)
        s[ty + p * 8][tx] = src[(size_t)(ty + p * 8) * HDIM + tx];
    __syncthreads();
    __bf16* dst = Wt + ((size_t)e * HDIM + d0) * HDIM + h0;
#pragma unroll
    for (int p = 0; p < 4; ++p)
        dst[(size_t)(ty + p * 8) * HDIM + tx] = (__bf16)s[tx][ty + p * 8];
}

// =====================================================================
// Kernel 2: LayerNorm -> bf16 z, router logits -> softmax -> top-2 weights
// One 256-thread block (8 wave32) per token.
// =====================================================================
__global__ void __launch_bounds__(256)
ln_router(const float* __restrict__ x, const float* __restrict__ g,
          const float* __restrict__ b, const float* __restrict__ Wr,
          const float* __restrict__ br, unsigned short* __restrict__ zh_,
          float* __restrict__ wout)
{
    __bf16* zh = reinterpret_cast<__bf16*>(zh_);
    const int t   = blockIdx.x;
    const int tid = threadIdx.x;
    const float4* xr = reinterpret_cast<const float4*>(x + (size_t)t * HDIM);

    float4 xv[3];
    float sum = 0.f, ss = 0.f;
#pragma unroll
    for (int i = 0; i < 3; ++i) {
        xv[i] = xr[tid + i * 256];
        sum += xv[i].x + xv[i].y + xv[i].z + xv[i].w;
        ss  += xv[i].x * xv[i].x + xv[i].y * xv[i].y +
               xv[i].z * xv[i].z + xv[i].w * xv[i].w;
    }
#pragma unroll
    for (int off = 16; off >= 1; off >>= 1) {
        sum += __shfl_xor(sum, off, 32);
        ss  += __shfl_xor(ss, off, 32);
    }
    __shared__ float red[8][2];
    __shared__ float ra[8][4];
    const int wave = tid >> 5, lane = tid & 31;
    if (lane == 0) { red[wave][0] = sum; red[wave][1] = ss; }
    __syncthreads();
    sum = 0.f; ss = 0.f;
#pragma unroll
    for (int i = 0; i < 8; ++i) { sum += red[i][0]; ss += red[i][1]; }
    const float mu  = sum * (1.0f / HDIM);
    const float var = ss * (1.0f / HDIM) - mu * mu;
    const float rs  = rsqrtf(var + 1e-5f);

    float a4[EXPERTS] = {0.f, 0.f, 0.f, 0.f};
    const float4* g4  = reinterpret_cast<const float4*>(g);
    const float4* b4  = reinterpret_cast<const float4*>(b);
    const float4* wr4 = reinterpret_cast<const float4*>(Wr);   // Wr row h = 4 experts
    v4bf* zrow = reinterpret_cast<v4bf*>(zh + (size_t)t * HDIM);

#pragma unroll
    for (int i = 0; i < 3; ++i) {
        const int j = tid + i * 256;
        const float4 gv = g4[j], bv = b4[j];
        float zc[4];
        zc[0] = (xv[i].x - mu) * rs * gv.x + bv.x;
        zc[1] = (xv[i].y - mu) * rs * gv.y + bv.y;
        zc[2] = (xv[i].z - mu) * rs * gv.z + bv.z;
        zc[3] = (xv[i].w - mu) * rs * gv.w + bv.w;
        v4bf p;
        p[0] = (__bf16)zc[0]; p[1] = (__bf16)zc[1];
        p[2] = (__bf16)zc[2]; p[3] = (__bf16)zc[3];
        zrow[j] = p;
#pragma unroll
        for (int c = 0; c < 4; ++c) {
            const float4 wv = wr4[4 * j + c];
            a4[0] += zc[c] * wv.x; a4[1] += zc[c] * wv.y;
            a4[2] += zc[c] * wv.z; a4[3] += zc[c] * wv.w;
        }
    }
#pragma unroll
    for (int off = 16; off >= 1; off >>= 1)
#pragma unroll
        for (int e = 0; e < EXPERTS; ++e) a4[e] += __shfl_xor(a4[e], off, 32);
    if (lane == 0)
#pragma unroll
        for (int e = 0; e < EXPERTS; ++e) ra[wave][e] = a4[e];
    __syncthreads();
    if (tid == 0) {
        float lg[EXPERTS];
#pragma unroll
        for (int e = 0; e < EXPERTS; ++e) {
            lg[e] = br[e];
            for (int i = 0; i < 8; ++i) lg[e] += ra[i][e];
        }
        const float mx = fmaxf(fmaxf(lg[0], lg[1]), fmaxf(lg[2], lg[3]));
        float p[EXPERTS], s = 0.f;
#pragma unroll
        for (int e = 0; e < EXPERTS; ++e) { p[e] = __expf(lg[e] - mx); s += p[e]; }
        const float inv = 1.f / s;
#pragma unroll
        for (int e = 0; e < EXPERTS; ++e) p[e] *= inv;
        int i1 = 0;
#pragma unroll
        for (int e = 1; e < EXPERTS; ++e) if (p[e] > p[i1]) i1 = e;
        int i2 = -1;
#pragma unroll
        for (int e = 0; e < EXPERTS; ++e)
            if (e != i1 && (i2 < 0 || p[e] > p[i2])) i2 = e;
        const float den = fmaxf(p[i1] + p[i2], 1e-8f);
#pragma unroll
        for (int e = 0; e < EXPERTS; ++e)
            wout[t * EXPERTS + e] = (e == i1) ? p[i1] / den
                                  : (e == i2) ? p[i2] / den : 0.f;
    }
}

// =====================================================================
// Kernel 3: dense all-expert GEMM + top-2 weighted combine + residual.
// Block = 128x128 tile, 8 wave32; wave = 32x64 subtile = 2x4 wmma accs.
// out[t,d] = x[t,d] + sum_e w[t,e] * ( z @ Wt[e]^T + be[e] )
// =====================================================================
__global__ void __launch_bounds__(256)
moe_gemm(const float* __restrict__ x, const unsigned short* __restrict__ zh_,
         const float* __restrict__ wgt, const unsigned short* __restrict__ Wt_,
         const float* __restrict__ be, float* __restrict__ out)
{
    const __bf16* zh = reinterpret_cast<const __bf16*>(zh_);
    const __bf16* Wt = reinterpret_cast<const __bf16*>(Wt_);

    __shared__ alignas(16) __bf16 As[2][TM * LDS_STRIDE];
    __shared__ alignas(16) __bf16 Bs[2][TN * LDS_STRIDE];
    __shared__ float wl[TM * EXPERTS];

    const int tid = threadIdx.x;
    const int t0  = blockIdx.y * TM;
    const int d0  = blockIdx.x * TN;

    for (int i = tid; i < TM * EXPERTS; i += 256)
        wl[i] = wgt[t0 * EXPERTS + i];

    const int wave = tid >> 5, lane = tid & 31;
    const int wr = wave & 3;     // 4 row-waves * 32 rows
    const int wc = wave >> 2;    // 2 col-waves * 64 cols
    const int l16 = lane & 15, hi = lane >> 4;

    v8f mixed[2][4] = {};

    for (int e = 0; e < EXPERTS; ++e) {
        const __bf16* Bgl = Wt + ((size_t)e * HDIM + d0) * HDIM;  // row = out col d, contiguous in k
        v8f acc[2][4] = {};

        // stage one 128x32 A tile and 128(x-col)x32 B tile into LDS buffer `buf`
        auto stage = [&](int buf, int k0) {
#pragma unroll
            for (int i = 0; i < 2; ++i) {
                const int lin = tid + i * 256;
                const int r = lin >> 2;
                const int c = (lin & 3) * 8;
#if USE_ASYNC_LDS
                // memory -> LDS without VGPR round-trip; tracked by ASYNCcnt
                __builtin_amdgcn_global_load_async_to_lds_b128(
                    (as1_v4i*)&zh[(size_t)(t0 + r) * HDIM + k0 + c],
                    (as3_v4i*)&As[buf][r * LDS_STRIDE + c], 0, 0);
                __builtin_amdgcn_global_load_async_to_lds_b128(
                    (as1_v4i*)&Bgl[(size_t)r * HDIM + k0 + c],
                    (as3_v4i*)&Bs[buf][r * LDS_STRIDE + c], 0, 0);
#else
                *reinterpret_cast<v8bf*>(&As[buf][r * LDS_STRIDE + c]) =
                    *reinterpret_cast<const v8bf*>(&zh[(size_t)(t0 + r) * HDIM + k0 + c]);
                *reinterpret_cast<v8bf*>(&Bs[buf][r * LDS_STRIDE + c]) =
                    *reinterpret_cast<const v8bf*>(&Bgl[(size_t)r * HDIM + k0 + c]);
#endif
            }
        };

        __syncthreads();   // buffers free from previous expert's compute
        stage(0, 0);

        constexpr int NK = HDIM / TK;   // 96
        for (int kt = 0; kt < NK; ++kt) {
            const int cur = kt & 1, nxt = cur ^ 1;
#if USE_ASYNC_LDS
            __builtin_amdgcn_s_wait_asynccnt(0);  // this thread's copies for `cur` landed
#endif
            __syncthreads();                       // everyone's copies landed; `nxt` free
            if (kt + 1 < NK) stage(nxt, (kt + 1) * TK);

            // A fragments: lane(l16) = row; lanes 0-15 hold K {0..7,16..23},
            // lanes 16-31 hold K {8..15,24..31} (ISA 16-bit A 16x32 layout).
            v16bf a[2], bfr[4];
#pragma unroll
            for (int mt = 0; mt < 2; ++mt) {
                const int row = wr * 32 + mt * 16 + l16;
                const v8bf lo =
                    *reinterpret_cast<const v8bf*>(&As[cur][row * LDS_STRIDE + hi * 8]);
                const v8bf h8 =
                    *reinterpret_cast<const v8bf*>(&As[cur][row * LDS_STRIDE + hi * 8 + 16]);
                a[mt] = __builtin_shufflevector(lo, h8, 0, 1, 2, 3, 4, 5, 6, 7,
                                                8, 9, 10, 11, 12, 13, 14, 15);
            }
            // B fragments: lane(l16) = out col; lanes 0-15 hold K {0..7,16..23},
            // lanes 16-31 hold K {8..15,24..31}; Wt rows are contiguous in K.
#pragma unroll
            for (int nt = 0; nt < 4; ++nt) {
                const int n = wc * 64 + nt * 16 + l16;
                const v8bf lo =
                    *reinterpret_cast<const v8bf*>(&Bs[cur][n * LDS_STRIDE + hi * 8]);
                const v8bf h8 =
                    *reinterpret_cast<const v8bf*>(&Bs[cur][n * LDS_STRIDE + hi * 8 + 16]);
                bfr[nt] = __builtin_shufflevector(lo, h8, 0, 1, 2, 3, 4, 5, 6, 7,
                                                  8, 9, 10, 11, 12, 13, 14, 15);
            }
#pragma unroll
            for (int mt = 0; mt < 2; ++mt)
#pragma unroll
                for (int nt = 0; nt < 4; ++nt)
                    acc[mt][nt] = __builtin_amdgcn_wmma_f32_16x16x32_bf16(
                        false, a[mt], false, bfr[nt], (short)0, acc[mt][nt],
                        false, false);
        }

        // epilogue: mixed += w[t,e] * (acc + be[e,col])
#pragma unroll
        for (int nt = 0; nt < 4; ++nt) {
            const float bev = be[e * HDIM + d0 + wc * 64 + nt * 16 + l16];
#pragma unroll
            for (int mt = 0; mt < 2; ++mt)
#pragma unroll
                for (int j = 0; j < 8; ++j) {
                    const int row = wr * 32 + mt * 16 + hi * 8 + j;  // C layout: M = j (+8 hi lanes)
                    mixed[mt][nt][j] += wl[row * EXPERTS + e] * (acc[mt][nt][j] + bev);
                }
        }
    }

    // residual + writeout (each output element owned by exactly one lane)
#pragma unroll
    for (int mt = 0; mt < 2; ++mt)
#pragma unroll
        for (int j = 0; j < 8; ++j) {
            const int row = t0 + wr * 32 + mt * 16 + hi * 8 + j;
#pragma unroll
            for (int nt = 0; nt < 4; ++nt) {
                const int col = d0 + wc * 64 + nt * 16 + l16;
                const size_t gi = (size_t)row * HDIM + col;
                out[gi] = x[gi] + mixed[mt][nt][j];
            }
        }
}

// =====================================================================
// Host launcher
// =====================================================================
extern "C" void kernel_launch(void* const* d_in, const int* in_sizes, int n_in,
                              void* d_out, int out_size, void* d_ws, size_t ws_size,
                              hipStream_t stream)
{
    const float* x   = (const float*)d_in[0];
    const float* lng = (const float*)d_in[1];
    const float* lnb = (const float*)d_in[2];
    const float* Wr  = (const float*)d_in[3];
    const float* br  = (const float*)d_in[4];
    const float* We  = (const float*)d_in[5];
    const float* be  = (const float*)d_in[6];
    float* out = (float*)d_out;

    char* ws = (char*)d_ws;
    const size_t WT_BYTES = (size_t)EXPERTS * HDIM * HDIM * sizeof(unsigned short); // 75.5 MB
    const size_t ZH_BYTES = (size_t)NTOK * HDIM * sizeof(unsigned short);           // 50.3 MB
    unsigned short* Wt = (unsigned short*)ws;
    unsigned short* zh = (unsigned short*)(ws + WT_BYTES);
    float*          w  = (float*)(ws + WT_BYTES + ZH_BYTES);

    for (int l = 0; l < LAYERS; ++l) {
        const float* xin = (l == 0) ? x : out;
        cvt_transpose_bf16<<<dim3(HDIM / 32, HDIM / 32, EXPERTS), 256, 0, stream>>>(
            We + (size_t)l * EXPERTS * HDIM * HDIM, Wt);
        ln_router<<<NTOK, 256, 0, stream>>>(
            xin, lng + (size_t)l * HDIM, lnb + (size_t)l * HDIM,
            Wr + (size_t)l * HDIM * EXPERTS, br + (size_t)l * EXPERTS, zh, w);
        moe_gemm<<<dim3(HDIM / TN, NTOK / TM), 256, 0, stream>>>(
            xin, zh, w, Wt, be + (size_t)l * EXPERTS * HDIM, out);
    }
}
